// FlaxArcticMoE_6897717477991
// MI455X (gfx1250) — matrix operand order repaired and checked by reading, not compiled, not run on hardware
//
#include <hip/hip_runtime.h>
#include <hip/hip_bf16.h>

typedef __attribute__((ext_vector_type(16))) __bf16 v16bf;
typedef __attribute__((ext_vector_type(8)))  float  v8f;

namespace {
constexpr int TOK = 4096;    // B*S tokens
constexpr int HD  = 2048;    // hidden dim
constexpr int FD  = 4096;    // ffn dim
constexpr int NE  = 8;       // experts
constexpr int BM  = 128;     // token tile
constexpr int BN  = 64;      // output-col tile
constexpr int BK  = 32;      // k tile (bf16 wmma K)
constexpr int LDA  = BK + 8; // A tile row stride: 40 elems = 80B (16B aligned)
constexpr int LDBT = BK + 8; // transposed-B tile row stride: 40 elems = 80B
}

__device__ __forceinline__ v8f wmma_bf16(v16bf a, v16bf b, v8f c) {
  return __builtin_amdgcn_wmma_f32_16x16x32_bf16(false, a, false, b, (short)0, c,
                                                 false, false);
}

// A fragment 16x32 bf16 (ISA 7.12.2): lane l holds row m=l&15,
// K = kHi..kHi+7 and 16+kHi..16+kHi+7 (kHi = 8*(l>=16)) -> two contiguous
// 16B chunks of the row-major LDS row => two ds_load_b128.
__device__ __forceinline__ v16bf load_frag_a(const __bf16* base, int mBase,
                                             int lane) {
  const __bf16* row = base + (mBase + (lane & 15)) * LDA + ((lane >> 4) << 3);
  union { uint4 q[2]; v16bf v; } u;
  u.q[0] = *(const uint4*)row;
  u.q[1] = *(const uint4*)(row + 16);
  return u.v;
}

// B fragment 32x16 bf16: lane l holds col n=l&15, K = kHi..kHi+7 and
// 16+kHi..16+kHi+7 (kHi = 8*(l>=16)). Tile stored TRANSPOSED Bt[n][k] so
// these are two contiguous 16B chunks => two ds_load_b128.
__device__ __forceinline__ v16bf load_frag_b(const __bf16* baseT, int nBase,
                                             int lane) {
  const __bf16* row = baseT + (nBase + (lane & 15)) * LDBT + ((lane >> 4) << 3);
  union { uint4 q[2]; v16bf v; } u;
  u.q[0] = *(const uint4*)row;
  u.q[1] = *(const uint4*)(row + 16);
  return u.v;
}

__global__ void zero_out_kernel(float* __restrict__ out, int n,
                                float* __restrict__ probsum) {
  int i = blockIdx.x * blockDim.x + threadIdx.x;
  if (i < n) out[i] = 0.0f;
  if (i < NE) probsum[i] = 0.0f;
}

__global__ void convert_x_kernel(const float* __restrict__ x,
                                 __bf16* __restrict__ xb, int n) {
  for (int i = blockIdx.x * blockDim.x + threadIdx.x; i < n;
       i += gridDim.x * blockDim.x)
    xb[i] = (__bf16)x[i];
}

// One wave32 per token: logits, top-2 + softmax, full softmax prob sums.
__global__ void router_kernel(const float* __restrict__ X,
                              const float* __restrict__ GW,
                              int* __restrict__ sel_e, float* __restrict__ sel_w,
                              float* __restrict__ probsum) {
  const int lane = threadIdx.x & 31;
  const int wave = threadIdx.x >> 5;
  const int t = blockIdx.x * 8 + wave;
  float acc[NE];
#pragma unroll
  for (int e = 0; e < NE; ++e) acc[e] = 0.0f;
  const float* xrow = X + (size_t)t * HD;
  for (int h = lane; h < HD; h += 32) {
    float xv = xrow[h];
    const float* g = GW + (size_t)h * NE;
#pragma unroll
    for (int e = 0; e < NE; ++e) acc[e] += xv * g[e];
  }
#pragma unroll
  for (int e = 0; e < NE; ++e) {
#pragma unroll
    for (int off = 16; off >= 1; off >>= 1)
      acc[e] += __shfl_xor(acc[e], off, 32);
  }
  if (lane == 0) {
    int e0 = 0;
    float l0 = acc[0];
#pragma unroll
    for (int e = 1; e < NE; ++e)
      if (acc[e] > l0) { l0 = acc[e]; e0 = e; }  // strict >: jax low-index tie-break
    int e1 = -1;
    float l1 = -3.402823466e38f;
#pragma unroll
    for (int e = 0; e < NE; ++e)
      if (e != e0 && acc[e] > l1) { l1 = acc[e]; e1 = e; }
    float w0 = 1.0f / (1.0f + __expf(l1 - l0));  // softmax over top-2
    sel_e[2 * t] = e0; sel_e[2 * t + 1] = e1;
    sel_w[2 * t] = w0; sel_w[2 * t + 1] = 1.0f - w0;
    float p[NE];
    float se = 0.0f;
#pragma unroll
    for (int e = 0; e < NE; ++e) { p[e] = __expf(acc[e] - l0); se += p[e]; }
    float inv = 1.0f / se;
#pragma unroll
    for (int e = 0; e < NE; ++e) atomicAdd(&probsum[e], p[e] * inv);
  }
}

// Deterministic compaction: thread e scans tokens in order.
__global__ void build_lists_kernel(const int* __restrict__ sel_e,
                                   const float* __restrict__ sel_w,
                                   int* __restrict__ list, float* __restrict__ cw,
                                   int* __restrict__ cnt) {
  int e = threadIdx.x;
  if (e >= NE) return;
  int c = 0;
  for (int t = 0; t < TOK; ++t) {
    if (sel_e[2 * t] == e)     { list[e * TOK + c] = t; cw[e * TOK + c] = sel_w[2 * t];     ++c; }
    if (sel_e[2 * t + 1] == e) { list[e * TOK + c] = t; cw[e * TOK + c] = sel_w[2 * t + 1]; ++c; }
  }
  cnt[e] = c;
}

// Load one 32x64 fp32 weight tile, convert to bf16, store TRANSPOSED to LDS.
// Thread owns (n, k-octet): 8 coalesced global b32 loads (lanes span n),
// one 16B LDS store. Weights stay resident in the 192MB L2 across M tiles.
__device__ __forceinline__ void load_btile_t(const float* __restrict__ colBase,
                                             size_t rowStride, int k0, int bn,
                                             int bko, __bf16* BtBase) {
  union { uint4 q; __bf16 e[8]; } u;
#pragma unroll
  for (int j = 0; j < 8; ++j)
    u.e[j] = (__bf16)colBase[(size_t)(k0 + bko + j) * rowStride];
  *(uint4*)(BtBase + bn * LDBT + bko) = u.q;
}

// Stage A: G[rows,F] = silu(Xg @ w1) * (Xg @ w3), rows gathered per expert.
// Double-buffered LDS: one barrier per k-step; next tile's global loads and
// LDS stores overlap the current tile's WMMA chain.
__global__ __launch_bounds__(256) void moe_stage_a(
    const __bf16* __restrict__ Xb, const float* __restrict__ W1,
    const float* __restrict__ W3, const int* __restrict__ list,
    const int* __restrict__ cntArr, __bf16* __restrict__ G, int expert) {
  const int cnt = cntArr[expert];
  const int mTile = blockIdx.y;
  if (mTile * BM >= cnt) return;
  const int n0 = blockIdx.x * BN;

  __shared__ __bf16 As[2][BM][LDA];
  __shared__ __bf16 B1t[2][BN][LDBT];
  __shared__ __bf16 B3t[2][BN][LDBT];

  const int tid = threadIdx.x;
  const int lane = tid & 31;
  const int wave = tid >> 5;

  const int ar = tid >> 1;          // A tile row 0..127
  const int aseg = (tid & 1) * 16;  // 16-elem (32B) segment
  int tok = -1;
  {
    int gr = mTile * BM + ar;
    if (gr < cnt) tok = list[expert * TOK + gr];
  }
  if (tok < 0) {  // zero both buffers once; region never rewritten below
    uint4 z = {0u, 0u, 0u, 0u};
    uint4* d0 = (uint4*)&As[0][ar][aseg];
    uint4* d1 = (uint4*)&As[1][ar][aseg];
    d0[0] = z; d0[1] = z; d1[0] = z; d1[1] = z;
  }

  const int bn = tid & 63;          // tile col n
  const int bko = (tid >> 6) << 3;  // k octet base: 0,8,16,24
  const float* w1col = W1 + ((size_t)expert * HD) * FD + n0 + bn;
  const float* w3col = W3 + ((size_t)expert * HD) * FD + n0 + bn;

  auto load_tile = [&](int k0, int buf) {
    if (tok >= 0) {
      const uint4* src = (const uint4*)(Xb + (size_t)tok * HD + k0 + aseg);
      uint4* dst = (uint4*)&As[buf][ar][aseg];
      dst[0] = src[0]; dst[1] = src[1];
    }
    load_btile_t(w1col, FD, k0, bn, bko, &B1t[buf][0][0]);
    load_btile_t(w3col, FD, k0, bn, bko, &B3t[buf][0][0]);
  };

  v8f acc1[4] = {};
  v8f acc3[4] = {};

  load_tile(0, 0);
  int buf = 0;
  for (int k0 = 0; k0 < HD; k0 += BK) {
    __syncthreads();
    if (k0 + BK < HD) load_tile(k0 + BK, buf ^ 1);
    // Stage all fragments, then back-to-back WMMAs (waits amortize).
    v16bf a = load_frag_a(&As[buf][0][0], wave * 16, lane);
    v16bf bf[8];
#pragma unroll
    for (int nt = 0; nt < 4; ++nt) {
      bf[2 * nt]     = load_frag_b(&B1t[buf][0][0], nt * 16, lane);
      bf[2 * nt + 1] = load_frag_b(&B3t[buf][0][0], nt * 16, lane);
    }
#pragma unroll
    for (int nt = 0; nt < 4; ++nt) {
      acc1[nt] = wmma_bf16(a, bf[2 * nt], acc1[nt]);
      acc3[nt] = wmma_bf16(a, bf[2 * nt + 1], acc3[nt]);
    }
    buf ^= 1;
  }

  // epilogue: fused silu(h1)*h3 -> bf16 G (C/D layout: VGPR r = rows r / r+8)
  const int mloc = (lane >> 4) << 3;
  const int ncol = n0 + (lane & 15);
  const int rowBase = mTile * BM + wave * 16 + mloc;
#pragma unroll
  for (int nt = 0; nt < 4; ++nt) {
#pragma unroll
    for (int r = 0; r < 8; ++r) {
      float h1 = acc1[nt][r];
      float h3 = acc3[nt][r];
      float g = (h1 / (1.0f + __expf(-h1))) * h3;
      G[(size_t)(rowBase + r) * FD + ncol + nt * 16] = (__bf16)g;
    }
  }
}

// Stage B: out[t,:] += combine * (G @ w2); same double-buffered scheme.
__global__ __launch_bounds__(256) void moe_stage_b(
    const __bf16* __restrict__ G, const float* __restrict__ W2,
    const int* __restrict__ list, const float* __restrict__ cw,
    const int* __restrict__ cntArr, float* __restrict__ out, int expert) {
  const int cnt = cntArr[expert];
  const int mTile = blockIdx.y;
  if (mTile * BM >= cnt) return;
  const int n0 = blockIdx.x * BN;

  __shared__ __bf16 As[2][BM][LDA];
  __shared__ __bf16 B2t[2][BN][LDBT];

  const int tid = threadIdx.x;
  const int lane = tid & 31;
  const int wave = tid >> 5;
  const int ar = tid >> 1;
  const int aseg = (tid & 1) * 16;
  const int bn = tid & 63;
  const int bko = (tid >> 6) << 3;

  const __bf16* arow = G + (size_t)(mTile * BM + ar) * FD + aseg;
  const float* w2col = W2 + ((size_t)expert * FD) * HD + n0 + bn;

  auto load_tile = [&](int k0, int buf) {
    const uint4* src = (const uint4*)(arow + k0);
    uint4* dst = (uint4*)&As[buf][ar][aseg];
    dst[0] = src[0]; dst[1] = src[1];
    load_btile_t(w2col, HD, k0, bn, bko, &B2t[buf][0][0]);
  };

  v8f acc[4] = {};

  load_tile(0, 0);
  int buf = 0;
  for (int k0 = 0; k0 < FD; k0 += BK) {
    __syncthreads();
    if (k0 + BK < FD) load_tile(k0 + BK, buf ^ 1);
    v16bf a = load_frag_a(&As[buf][0][0], wave * 16, lane);
    v16bf bf[4];
#pragma unroll
    for (int nt = 0; nt < 4; ++nt)
      bf[nt] = load_frag_b(&B2t[buf][0][0], nt * 16, lane);
#pragma unroll
    for (int nt = 0; nt < 4; ++nt) acc[nt] = wmma_bf16(a, bf[nt], acc[nt]);
    buf ^= 1;
  }

  const int mloc = (lane >> 4) << 3;
  const int ncol = n0 + (lane & 15);
#pragma unroll
  for (int nt = 0; nt < 4; ++nt) {
#pragma unroll
    for (int r = 0; r < 8; ++r) {
      int gr = mTile * BM + wave * 16 + mloc + r;
      if (gr < cnt) {
        int t = list[expert * TOK + gr];
        float wgt = cw[expert * TOK + gr];
        atomicAdd(&out[(size_t)t * HD + ncol + nt * 16], wgt * acc[nt][r]);
      }
    }
  }
}

// aux = E * sum_e mean_prob_e * (count_e / T); counts over (t,k) pairs, /T again
__global__ void aux_final_kernel(const float* __restrict__ probsum,
                                 const int* __restrict__ cnt,
                                 float* __restrict__ aux_out) {
  if (threadIdx.x == 0 && blockIdx.x == 0) {
    float s = 0.0f;
    for (int e = 0; e < NE; ++e)
      s += (probsum[e] / (float)TOK) * ((float)cnt[e] / (float)TOK);
    *aux_out = s * (float)NE;
  }
}

extern "C" void kernel_launch(void* const* d_in, const int* in_sizes, int n_in,
                              void* d_out, int out_size, void* d_ws,
                              size_t ws_size, hipStream_t stream) {
  (void)in_sizes; (void)n_in; (void)ws_size;
  const float* X  = (const float*)d_in[0];   // [B,S,H] f32
  const float* GW = (const float*)d_in[1];   // [H,E]   f32
  const float* W1 = (const float*)d_in[2];   // [E,H,F] f32
  const float* W3 = (const float*)d_in[3];   // [E,H,F] f32
  const float* W2 = (const float*)d_in[4];   // [E,F,H] f32
  float* out = (float*)d_out;                // final [B,S,H] ++ aux (1)

  size_t off = 0;
  char* ws = (char*)d_ws;
  auto take = [&](size_t bytes) -> char* {
    off = (off + 255) & ~(size_t)255;
    char* p = ws + off;
    off += bytes;
    return p;
  };
  float*  probsum = (float*)take(NE * sizeof(float));
  int*    cnt     = (int*)take(NE * sizeof(int));
  int*    sel_e   = (int*)take((size_t)2 * TOK * sizeof(int));
  float*  sel_w   = (float*)take((size_t)2 * TOK * sizeof(float));
  int*    list    = (int*)take((size_t)NE * TOK * sizeof(int));
  float*  cwp     = (float*)take((size_t)NE * TOK * sizeof(float));
  __bf16* Xb      = (__bf16*)take((size_t)TOK * HD * sizeof(__bf16));
  __bf16* G       = (__bf16*)take((size_t)TOK * FD * sizeof(__bf16));

  zero_out_kernel<<<(out_size + 255) / 256, 256, 0, stream>>>(out, out_size,
                                                              probsum);
  convert_x_kernel<<<1024, 256, 0, stream>>>(X, Xb, TOK * HD);
  router_kernel<<<TOK / 8, 256, 0, stream>>>(X, GW, sel_e, sel_w, probsum);
  build_lists_kernel<<<1, 32, 0, stream>>>(sel_e, sel_w, list, cwp, cnt);

  for (int e = 0; e < NE; ++e) {
    moe_stage_a<<<dim3(FD / BN, TOK / BM), 256, 0, stream>>>(Xb, W1, W3, list,
                                                             cnt, G, e);
    moe_stage_b<<<dim3(HD / BN, TOK / BM), 256, 0, stream>>>(G, W2, list, cwp,
                                                             cnt, out, e);
  }
  aux_final_kernel<<<1, 1, 0, stream>>>(probsum, cnt, out + (out_size - 1));
}
